// MultiHeadAttention_188978561088
// MI455X (gfx1250) — matrix operand order repaired and checked by reading, not compiled, not run on hardware
//
#include <hip/hip_runtime.h>
#include <hip/hip_bf16.h>
#include <math.h>

#define S_LEN  4096
#define DMODEL 1024
#define NHEADS 16
#define DHEAD  64

typedef __attribute__((ext_vector_type(16))) __bf16       v16bf;
typedef __attribute__((ext_vector_type(8)))  __bf16       v8bf;
typedef __attribute__((ext_vector_type(8)))  float        v8f;
typedef __attribute__((ext_vector_type(8)))  unsigned int v8u;
typedef __attribute__((ext_vector_type(4)))  unsigned int v4u;

static __device__ __forceinline__ v8f wmma_bf16(v16bf a, v16bf b, v8f c) {
  // D(f32 16x16) = A(16x32 bf16) * B(32x16 bf16) + C
  return __builtin_amdgcn_wmma_f32_16x16x32_bf16(false, a, false, b, (short)0, c,
                                                 false, false);
}

static __device__ __forceinline__ unsigned short f2bf_sw(float f) {
  unsigned int u = __float_as_uint(f);
  u += 0x7fffu + ((u >> 16) & 1u);   // round-to-nearest-even
  return (unsigned short)(u >> 16);
}

// Packed f32x2 -> bf16x2 (low = a, high = b). Hardware v_cvt_pk_bf16_f32 when
// available, manual RNE otherwise.
#if __has_builtin(__builtin_amdgcn_cvt_pk_bf16_f32)
static __device__ __forceinline__ unsigned int pk_bf16(float a, float b) {
  auto r = __builtin_amdgcn_cvt_pk_bf16_f32(a, b);
  return __builtin_bit_cast(unsigned int, r);
}
#else
static __device__ __forceinline__ unsigned int pk_bf16(float a, float b) {
  return (unsigned int)f2bf_sw(a) | ((unsigned int)f2bf_sw(b) << 16);
}
#endif

static __device__ __forceinline__ v8f v8f_zero() {
  v8f z = {0.f, 0.f, 0.f, 0.f, 0.f, 0.f, 0.f, 0.f};
  return z;
}

// ---------------------------------------------------------------- converts
// Two f32 -> one packed bf16x2 dword per thread.
__global__ void cvt_bf16_kernel(const float2* __restrict__ in,
                                unsigned int* __restrict__ out, int npair) {
  int i = blockIdx.x * blockDim.x + threadIdx.x;
  if (i < npair) {
    float2 v = in[i];
    out[i] = pk_bf16(v.x, v.y);
  }
}

// ---------------------------------------------------------------- GEMM
// C[M][N] = A[M][K] * B[N][K]^T   (both operands K-contiguous in memory)
// Wave tile 32x64 (8 WMMAs per k-step, 2 A-tiles + 4 B-tiles loaded).
// Block: 128 threads = 4 waves as 2x2 -> block tile 64x128.
// mode 0: store f32 row-major [M][N]
// mode 1: store bf16 transposed [N][M]   (used for V -> Vt)
__global__ __launch_bounds__(128)
void gemm_bf16_kernel(const unsigned short* __restrict__ A,
                      const unsigned short* __restrict__ B,
                      void* __restrict__ Cout,
                      int M, int N, int K, int mode) {
  const int lane = threadIdx.x & 31;
  const int wave = threadIdx.x >> 5;
  const int lr = lane & 15;
  const int lh = lane >> 4;
  const int bm = blockIdx.y * 64 + (wave >> 1) * 32;
  const int bn = blockIdx.x * 128 + (wave & 1) * 64;

  v8f acc[2][4];
#pragma unroll
  for (int i = 0; i < 2; ++i)
#pragma unroll
    for (int j = 0; j < 4; ++j) acc[i][j] = v8f_zero();

  const unsigned short* arow[2] = { A + (size_t)(bm + lr) * K,
                                    A + (size_t)(bm + 16 + lr) * K };
  const unsigned short* brow[4] = { B + (size_t)(bn + lr) * K,
                                    B + (size_t)(bn + 16 + lr) * K,
                                    B + (size_t)(bn + 32 + lr) * K,
                                    B + (size_t)(bn + 48 + lr) * K };

  for (int k = 0; k < K; k += 32) {
    __builtin_prefetch(arow[0] + k + 256, 0, 1);
    __builtin_prefetch(brow[0] + k + 256, 0, 1);
    v16bf a[2], b[4];
#pragma unroll
    for (int t = 0; t < 2; ++t) {
      // A layout: lanes 0-15 hold K=0..15, lanes 16-31 K=16..31
      a[t] = *(const v16bf*)(arow[t] + k + lh * 16);
    }
#pragma unroll
    for (int t = 0; t < 4; ++t)
      // B layout: lane = column, lanes 0-15 K=0..15, lanes 16-31 K=16..31
      b[t] = *(const v16bf*)(brow[t] + k + lh * 16);
#pragma unroll
    for (int i = 0; i < 2; ++i)
#pragma unroll
      for (int j = 0; j < 4; ++j) acc[i][j] = wmma_bf16(a[i], b[j], acc[i][j]);
  }

  if (mode == 0) {
    float* C = (float*)Cout;
#pragma unroll
    for (int i = 0; i < 2; ++i)
#pragma unroll
      for (int j = 0; j < 4; ++j) {
        const int col = bn + j * 16 + lr;
        const int row0 = bm + i * 16 + lh * 8;   // C layout: vgpr r -> row row0+r
#pragma unroll
        for (int r = 0; r < 8; ++r)
          C[(size_t)(row0 + r) * N + col] = acc[i][j][r];
      }
  } else {
    unsigned short* C = (unsigned short*)Cout;   // [N][M] bf16
#pragma unroll
    for (int i = 0; i < 2; ++i)
#pragma unroll
      for (int j = 0; j < 4; ++j) {
        const int col = bn + j * 16 + lr;
        const int row0 = bm + i * 16 + lh * 8;
        v4u pk;
#pragma unroll
        for (int r = 0; r < 4; ++r)
          pk[r] = pk_bf16(acc[i][j][2 * r], acc[i][j][2 * r + 1]);
        *(v4u*)(C + (size_t)col * M + row0) = pk;   // 8 consecutive rows, 16B
      }
  }
}

// ---------------------------------------------------------------- RoPE + bf16
// Reads f32 [S][1024] row-major, writes bf16 [H][S][64] head-major (scaled).
__global__ void rope_cvt_kernel(const float* __restrict__ X,
                                const int* __restrict__ pos,
                                unsigned short* __restrict__ Out, float scale) {
  int idx = blockIdx.x * blockDim.x + threadIdx.x;
  if (idx >= S_LEN * (DMODEL / 2)) return;
  const int s = idx >> 9;          // 512 pairs per token
  const int p = idx & 511;
  const int h = p >> 5;
  const int pp = p & 31;           // pair index within head (dh = 2*pp)
  const int o = h * 64 + 2 * pp;
  // inv_freq = theta^(-2*pp/64) = exp(-pp * ln(1e4)/32)
  float fr = __expf(-(float)pp * (9.210340371976184f / 32.0f));
  float ang = (float)pos[s] * fr;
  float sn, cs;
  __sincosf(ang, &sn, &cs);
  float xe = X[(size_t)s * DMODEL + o];
  float xo = X[(size_t)s * DMODEL + o + 1];
  float ye = (cs * xe - sn * xo) * scale;
  float yo = (sn * xe + cs * xo) * scale;
  *(unsigned int*)(Out + ((size_t)h * S_LEN + s) * DHEAD + 2 * pp) =
      pk_bf16(ye, yo);
}

// ---------------------------------------------------------------- flash attn
// One wave (32 threads) per (head, 16-query tile). Streams causal 32-key tiles:
// 4 score WMMAs + online softmax + P relayout through LDS + 4 ctx WMMAs.
// Qb pre-scaled by 1/sqrt(64). Writes bf16 ctx [S][1024].
__global__ __launch_bounds__(32)
void flash_attn_kernel(const unsigned short* __restrict__ Qb,   // [H][S][64]
                       const unsigned short* __restrict__ Kb,   // [H][S][64]
                       const unsigned short* __restrict__ Vt,   // [1024][S]
                       unsigned short* __restrict__ ctx) {      // [S][1024]
  const int h = blockIdx.y;
  const int q0 = blockIdx.x * 16;
  const int lane = threadIdx.x & 31;
  const int lr = lane & 15;
  const int lh = lane >> 4;

  __shared__ unsigned short pl[16][34];   // P tile, padded stride (17 banks)

  // Q as A operand, 2 k-steps over dh=64
  const unsigned short* qrow = Qb + ((size_t)h * S_LEN + q0 + lr) * DHEAD;
  v16bf qa[2];
#pragma unroll
  for (int t = 0; t < 2; ++t) {
    qa[t] = *(const v16bf*)(qrow + t * 32 + lh * 16);
  }

  v8f oacc[4];
#pragma unroll
  for (int t = 0; t < 4; ++t) oacc[t] = v8f_zero();
  float mrow[8], lrow[8];
#pragma unroll
  for (int i = 0; i < 8; ++i) { mrow[i] = -__builtin_inff(); lrow[i] = 0.f; }

  const int jmax = (q0 + 15) >> 5;   // causal: keys <= q0+15
  for (int j = 0; j <= jmax; ++j) {
    const int k0 = j * 32;
    // scores: two 16-key sub-tiles, K as B operand (lane = key, dh contiguous)
    v8f sc[2];
#pragma unroll
    for (int st = 0; st < 2; ++st) {
      const unsigned short* krow =
          Kb + ((size_t)h * S_LEN + k0 + st * 16 + lr) * DHEAD;
      v8f c = v8f_zero();
#pragma unroll
      for (int t = 0; t < 2; ++t)
        c = wmma_bf16(qa[t], *(const v16bf*)(krow + t * 32 + lh * 16), c);
      sc[st] = c;
    }
    // online softmax; C-layout row for vgpr i is (i + 8*lh); xor masks 1..8
    // stay inside each 16-lane half, matching that row split.
#pragma unroll
    for (int i = 0; i < 8; ++i) {
      const int qr = q0 + i + lh * 8;
      float s0 = ((k0 + lr) <= qr) ? sc[0][i] : -__builtin_inff();
      float s1 = ((k0 + 16 + lr) <= qr) ? sc[1][i] : -__builtin_inff();
      float mx = fmaxf(s0, s1);
      mx = fmaxf(mx, __shfl_xor(mx, 1));
      mx = fmaxf(mx, __shfl_xor(mx, 2));
      mx = fmaxf(mx, __shfl_xor(mx, 4));
      mx = fmaxf(mx, __shfl_xor(mx, 8));
      const float mnew = fmaxf(mrow[i], mx);
      const float alpha = __expf(mrow[i] - mnew);
      mrow[i] = mnew;
      const float p0 = __expf(s0 - mnew);
      const float p1 = __expf(s1 - mnew);
      float rs = p0 + p1;
      rs += __shfl_xor(rs, 1);
      rs += __shfl_xor(rs, 2);
      rs += __shfl_xor(rs, 4);
      rs += __shfl_xor(rs, 8);
      lrow[i] = lrow[i] * alpha + rs;
      const unsigned int p01 = pk_bf16(p0, p1);   // one v_cvt_pk_bf16_f32
      pl[i + lh * 8][lr] = (unsigned short)p01;
      pl[i + lh * 8][16 + lr] = (unsigned short)(p01 >> 16);
#pragma unroll
      for (int t = 0; t < 4; ++t) oacc[t][i] *= alpha;   // rescale running ctx
    }
    asm volatile("s_wait_dscnt 0" ::: "memory");   // in-wave LDS RAW
    // reload P in A-operand layout (16x32, K = key index)
    union { v8u u; v16bf b; } pu;
    const unsigned short* prow = &pl[lr][0];
#pragma unroll
    for (int v = 0; v < 8; ++v) {
      const int kb = 2 * v + lh * 16;
      pu.u[v] = *(const unsigned int*)(prow + kb);
    }
    const v16bf pa = pu.b;
    // ctx += P * V ; Vt rows (dh) are key-contiguous => direct B operand
#pragma unroll
    for (int dt = 0; dt < 4; ++dt) {
      const unsigned short* vrow =
          Vt + (size_t)(h * 64 + dt * 16 + lr) * S_LEN + k0 + lh * 16;
      oacc[dt] = wmma_bf16(pa, *(const v16bf*)vrow, oacc[dt]);
    }
  }
  // normalize and emit bf16 ctx [s][h*64+dh]
#pragma unroll
  for (int dt = 0; dt < 4; ++dt) {
    const int col = h * 64 + dt * 16 + lr;
#pragma unroll
    for (int i = 0; i < 8; ++i) {
      const int s = q0 + i + lh * 8;
      const float v = oacc[dt][i] * (1.0f / lrow[i]);
      ctx[(size_t)s * DMODEL + col] = (unsigned short)pk_bf16(v, v);
    }
  }
}

// ---------------------------------------------------------------- launch
extern "C" void kernel_launch(void* const* d_in, const int* in_sizes, int n_in,
                              void* d_out, int out_size, void* d_ws,
                              size_t ws_size, hipStream_t stream) {
  const float* x  = (const float*)d_in[0];
  const float* Wq = (const float*)d_in[1];
  const float* Wk = (const float*)d_in[2];
  const float* Wv = (const float*)d_in[3];
  const float* Wo = (const float*)d_in[4];
  const int* tpos = (const int*)d_in[5];
  float* out = (float*)d_out;

  char* ws = (char*)d_ws;
  const size_t MB = 1024u * 1024u;
  unsigned short* xb   = (unsigned short*)(ws + 0 * MB);    // 8 MB
  unsigned short* Wqb  = (unsigned short*)(ws + 8 * MB);    // 2 MB
  unsigned short* Wkb  = (unsigned short*)(ws + 10 * MB);
  unsigned short* Wvb  = (unsigned short*)(ws + 12 * MB);
  unsigned short* Wob  = (unsigned short*)(ws + 14 * MB);
  float*          Qf   = (float*)(ws + 16 * MB);            // 16 MB
  float*          Kf   = (float*)(ws + 32 * MB);            // 16 MB
  unsigned short* Qb   = (unsigned short*)(ws + 48 * MB);   // 8 MB
  unsigned short* Kb   = (unsigned short*)(ws + 56 * MB);   // 8 MB
  unsigned short* Vt   = (unsigned short*)(ws + 64 * MB);   // 8 MB -> 72 MB peak
  unsigned short* ctxb = (unsigned short*)(ws + 16 * MB);   // aliases dead Qf

  const int nxp = S_LEN * DMODEL / 2, nwp = DMODEL * DMODEL / 2;
  cvt_bf16_kernel<<<(nxp + 255) / 256, 256, 0, stream>>>(
      (const float2*)x, (unsigned int*)xb, nxp);
  cvt_bf16_kernel<<<(nwp + 255) / 256, 256, 0, stream>>>(
      (const float2*)Wq, (unsigned int*)Wqb, nwp);
  cvt_bf16_kernel<<<(nwp + 255) / 256, 256, 0, stream>>>(
      (const float2*)Wk, (unsigned int*)Wkb, nwp);
  cvt_bf16_kernel<<<(nwp + 255) / 256, 256, 0, stream>>>(
      (const float2*)Wv, (unsigned int*)Wvb, nwp);
  cvt_bf16_kernel<<<(nwp + 255) / 256, 256, 0, stream>>>(
      (const float2*)Wo, (unsigned int*)Wob, nwp);

  dim3 ggrid(DMODEL / 128, S_LEN / 64);   // (8, 64), 128 threads = 4 waves
  gemm_bf16_kernel<<<ggrid, 128, 0, stream>>>(xb, Wqb, Qf, S_LEN, DMODEL, DMODEL, 0);
  gemm_bf16_kernel<<<ggrid, 128, 0, stream>>>(xb, Wkb, Kf, S_LEN, DMODEL, DMODEL, 0);
  gemm_bf16_kernel<<<ggrid, 128, 0, stream>>>(xb, Wvb, Vt, S_LEN, DMODEL, DMODEL, 1);

  const int nr = S_LEN * (DMODEL / 2);
  rope_cvt_kernel<<<(nr + 255) / 256, 256, 0, stream>>>(Qf, tpos, Qb, 0.125f);
  rope_cvt_kernel<<<(nr + 255) / 256, 256, 0, stream>>>(Kf, tpos, Kb, 1.0f);

  flash_attn_kernel<<<dim3(S_LEN / 16, NHEADS), 32, 0, stream>>>(Qb, Kb, Vt, ctxb);

  gemm_bf16_kernel<<<ggrid, 128, 0, stream>>>(ctxb, Wob, out, S_LEN, DMODEL, DMODEL, 0);
}
